// QuantizedLinear_55911884259380
// MI455X (gfx1250) — 8-bit matrix operand order repaired and checked by reading, compiled, not run on hardware
//
#include <hip/hip_runtime.h>
#include <hip/hip_bf16.h>

typedef __attribute__((ext_vector_type(8))) int v8i;
typedef __attribute__((ext_vector_type(4))) int v4i_;

#define TOKENS 4096
#define IN_F   4096
#define OUT_F  4096
#define BM 128
#define BN 256
#define BK 64
#define NKTILES (IN_F / BK)

#if defined(__has_builtin)
#  if __has_builtin(__builtin_amdgcn_global_load_async_to_lds_b128)
#    define HAVE_ASYNC_BUILTIN 1
#  endif
#  if __has_builtin(__builtin_amdgcn_s_wait_asynccnt)
#    define HAVE_WAIT_ASYNC_BUILTIN 1
#  endif
#endif

typedef __attribute__((address_space(1))) v4i_* gv4_t;
typedef __attribute__((address_space(3))) v4i_* lv4_t;

__device__ __forceinline__ void async_copy16(const void* g, void* l) {
#if defined(HAVE_ASYNC_BUILTIN)
    __builtin_amdgcn_global_load_async_to_lds_b128((gv4_t)g, (lv4_t)l, 0, 0);
#else
    unsigned int loff =
        (unsigned int)(__UINTPTR_TYPE__)(__attribute__((address_space(3))) char*)l;
    asm volatile("global_load_async_to_lds_b128 %0, %1, off"
                 :: "v"(loff), "v"(g) : "memory");
#endif
}

__device__ __forceinline__ void wait_async_all() {
#if defined(HAVE_WAIT_ASYNC_BUILTIN)
    __builtin_amdgcn_s_wait_asynccnt(0);
#else
    asm volatile("s_wait_asynccnt 0x0" ::: "memory");
#endif
}

__global__ void init_ws(unsigned int* amax) {
    if (threadIdx.x == 0 && blockIdx.x == 0) *amax = 0u;
}

__global__ __launch_bounds__(256) void amax_kernel(const float4* __restrict__ x,
                                                   unsigned int* __restrict__ amax,
                                                   int n4) {
    __shared__ unsigned int red[256];
    float m = 0.0f;
    for (int i = blockIdx.x * blockDim.x + threadIdx.x; i < n4;
         i += gridDim.x * blockDim.x) {
        float4 v = x[i];
        m = fmaxf(m, fmaxf(fmaxf(fabsf(v.x), fabsf(v.y)),
                           fmaxf(fabsf(v.z), fabsf(v.w))));
    }
    red[threadIdx.x] = __float_as_uint(m);
    __syncthreads();
    for (int s = 128; s > 0; s >>= 1) {
        if (threadIdx.x < s) {
            unsigned int a = red[threadIdx.x], b = red[threadIdx.x + s];
            red[threadIdx.x] = a > b ? a : b;
        }
        __syncthreads();
    }
    if (threadIdx.x == 0) atomicMax(amax, red[0]);
}

__global__ __launch_bounds__(256) void quant_x(const float4* __restrict__ x,
                                               const unsigned int* __restrict__ amax,
                                               unsigned int* __restrict__ qx,
                                               int n4) {
    int i = blockIdx.x * blockDim.x + threadIdx.x;
    if (i >= n4) return;
    float a   = __uint_as_float(*amax);
    float inv = 127.0f / a;
    float4 v = x[i];
    int q0 = (int)rintf(v.x * inv);
    int q1 = (int)rintf(v.y * inv);
    int q2 = (int)rintf(v.z * inv);
    int q3 = (int)rintf(v.w * inv);
    q0 = q0 < -127 ? -127 : (q0 > 127 ? 127 : q0);
    q1 = q1 < -127 ? -127 : (q1 > 127 ? 127 : q1);
    q2 = q2 < -127 ? -127 : (q2 > 127 ? 127 : q2);
    q3 = q3 < -127 ? -127 : (q3 > 127 ? 127 : q3);
    qx[i] = (unsigned int)(q0 & 255) | ((unsigned int)(q1 & 255) << 8) |
            ((unsigned int)(q2 & 255) << 16) | ((unsigned int)(q3 & 255) << 24);
}

__global__ __launch_bounds__(256) void pack_w(const int4* __restrict__ w,
                                              unsigned int* __restrict__ qw,
                                              int n4) {
    int i = blockIdx.x * blockDim.x + threadIdx.x;
    if (i >= n4) return;
    int4 v = w[i];
    qw[i] = (unsigned int)(v.x & 255) | ((unsigned int)(v.y & 255) << 8) |
            ((unsigned int)(v.z & 255) << 16) | ((unsigned int)(v.w & 255) << 24);
}

__global__ __launch_bounds__(256) void gemm_i8(const uint4* __restrict__ qx,
                                               const uint4* __restrict__ qw,
                                               const unsigned int* __restrict__ amaxBits,
                                               const float* __restrict__ wscale,
                                               const float* __restrict__ bias,
                                               float* __restrict__ out) {
    __shared__ __align__(16) unsigned char lsA[2][BM * BK];
    __shared__ __align__(16) unsigned char lsB[2][BN * BK];

    const int tid  = threadIdx.x;
    const int lane = tid & 31;
    const int wave = tid >> 5;
    const int wm   = wave >> 2;
    const int wn   = wave & 3;
    const int rowBaseA = blockIdx.y * BM;
    const int rowBaseB = blockIdx.x * BN;

    const int g_row_u4 = IN_F / 16;
    const int half = lane >> 4;
    const int l16  = lane & 15;

    const int sRow[4] = { (tid + 0) >> 2, (tid + 256) >> 2,
                          (tid + 512) >> 2, (tid + 768) >> 2 };
    const int sC4[4]  = { (tid + 0) & 3, (tid + 256) & 3,
                          (tid + 512) & 3, (tid + 768) & 3 };

    v8i acc[4][4] = {};

    {
        async_copy16(&qx[(size_t)(rowBaseA + sRow[0]) * g_row_u4 + sC4[0]],
                     &lsA[0][(tid + 0) * 16]);
        async_copy16(&qx[(size_t)(rowBaseA + sRow[1]) * g_row_u4 + sC4[1]],
                     &lsA[0][(tid + 256) * 16]);
        #pragma unroll
        for (int r = 0; r < 4; ++r)
            async_copy16(&qw[(size_t)(rowBaseB + sRow[r]) * g_row_u4 + sC4[r]],
                         &lsB[0][(tid + r * 256) * 16]);
    }
    wait_async_all();
    __syncthreads();

    #pragma clang loop unroll(disable)
    for (int kb = 0; kb < NKTILES; ++kb) {
        const int cur = kb & 1;

        if (kb + 1 < NKTILES) {
            const int kU4 = (kb + 1) * (BK / 16);
            const int nxt = cur ^ 1;
            async_copy16(&qx[(size_t)(rowBaseA + sRow[0]) * g_row_u4 + kU4 + sC4[0]],
                         &lsA[nxt][(tid + 0) * 16]);
            async_copy16(&qx[(size_t)(rowBaseA + sRow[1]) * g_row_u4 + kU4 + sC4[1]],
                         &lsA[nxt][(tid + 256) * 16]);
            #pragma unroll
            for (int r = 0; r < 4; ++r)
                async_copy16(&qw[(size_t)(rowBaseB + sRow[r]) * g_row_u4 + kU4 + sC4[r]],
                             &lsB[nxt][(tid + r * 256) * 16]);
        }

        union FA { v8i v; unsigned long long q[4]; } fa[4];
        const unsigned long long* lA = (const unsigned long long*)lsA[cur];
        #pragma unroll
        for (int mt = 0; mt < 4; ++mt) {
            int row  = wm * 64 + mt * 16 + l16;
            int base = row * 8 + half * 2;
            fa[mt].q[0] = lA[base + 0];
            fa[mt].q[1] = lA[base + 1];
            fa[mt].q[2] = lA[base + 4];
            fa[mt].q[3] = lA[base + 5];
        }

        const uint4* lB = (const uint4*)lsB[cur];
        #pragma unroll
        for (int nt = 0; nt < 4; ++nt) {
            union FB { v8i v; uint4 u[2]; } fb;
            int row  = wn * 64 + nt * 16 + l16;
            int base = row * 4 + half;
            fb.u[0] = lB[base + 0];
            fb.u[1] = lB[base + 2];
            #pragma unroll
            for (int mt = 0; mt < 4; ++mt)
                acc[mt][nt] = __builtin_amdgcn_wmma_i32_16x16x64_iu8(
                    true, fa[mt].v, true, fb.v, acc[mt][nt], false, false);
        }

        wait_async_all();
        __syncthreads();
    }

    const float a = __uint_as_float(*amaxBits);
    const float s = (a / 127.0f) * wscale[0];
    #pragma unroll
    for (int nt = 0; nt < 4; ++nt) {
        int col = rowBaseB + wn * 64 + nt * 16 + l16;
        float b = bias[col];
        #pragma unroll
        for (int mt = 0; mt < 4; ++mt) {
            int rowB = rowBaseA + wm * 64 + mt * 16 + half * 8;
            #pragma unroll
            for (int j = 0; j < 8; ++j) {
                out[(size_t)(rowB + j) * OUT_F + col] = (float)acc[mt][nt][j] * s + b;
            }
        }
    }
}

extern "C" void kernel_launch(void* const* d_in, const int* in_sizes, int n_in,
                              void* d_out, int out_size, void* d_ws, size_t ws_size,
                              hipStream_t stream) {
    const float* x      = (const float*)d_in[0];
    const int*   w      = (const int*)d_in[1];
    const float* wscale = (const float*)d_in[2];
    const float* bias   = (const float*)d_in[3];
    float*       out    = (float*)d_out;

    unsigned char* ws   = (unsigned char*)d_ws;
    unsigned int*  amax = (unsigned int*)ws;
    unsigned int*  qx   = (unsigned int*)(ws + 256);
    unsigned int*  qw   = (unsigned int*)(ws + 256 + (size_t)TOKENS * IN_F);

    const int n  = TOKENS * IN_F;
    const int n4 = n / 4;

    init_ws<<<1, 64, 0, stream>>>(amax);
    amax_kernel<<<4096, 256, 0, stream>>>((const float4*)x, amax, n4);
    quant_x<<<n4 / 256, 256, 0, stream>>>((const float4*)x, amax, qx, n4);
    pack_w<<<n4 / 256, 256, 0, stream>>>((const int4*)w, qw, n4);

    dim3 grid(OUT_F / BN, TOKENS / BM);
    gemm_i8<<<grid, 256, 0, stream>>>((const uint4*)qx, (const uint4*)qw,
                                      amax, wscale, bias, out);
}
